// SelfAttn_54778012893327
// MI455X (gfx1250) — matrix operand order repaired and checked by reading, not compiled, not run on hardware
//
#include <hip/hip_runtime.h>
#include <hip/hip_bf16.h>

typedef __attribute__((ext_vector_type(16))) __bf16 v16bf;
typedef __attribute__((ext_vector_type(8)))  __bf16 v8bf;
typedef __attribute__((ext_vector_type(4)))  __bf16 v4bf;
typedef __attribute__((ext_vector_type(8)))  float  v8f;

#define EMB 1024
#define NH  16
#define HD  64
#define BB  2
#define SS  2048
#define MROWS (BB*SS)        // 4096
#define LOG2E 1.4426950408889634f

__device__ __forceinline__ v8bf ld8(const __bf16* p) { return *(const v8bf*)p; }
__device__ __forceinline__ v16bf join16(v8bf lo, v8bf hi) {
  return __builtin_shufflevector(lo, hi, 0,1,2,3,4,5,6,7,8,9,10,11,12,13,14,15);
}
__device__ __forceinline__ v8f wmma_bf16(v16bf a, v16bf b, v8f c) {
  return __builtin_amdgcn_wmma_f32_16x16x32_bf16(false, a, false, b, (short)0, c, false, false);
}
__device__ __forceinline__ float rmax16(float t) {
  t = fmaxf(t, __shfl_xor(t, 1, 16));
  t = fmaxf(t, __shfl_xor(t, 2, 16));
  t = fmaxf(t, __shfl_xor(t, 4, 16));
  t = fmaxf(t, __shfl_xor(t, 8, 16));
  return t;
}
__device__ __forceinline__ float rsum16(float t) {
  t += __shfl_xor(t, 1, 16);
  t += __shfl_xor(t, 2, 16);
  t += __shfl_xor(t, 4, 16);
  t += __shfl_xor(t, 8, 16);
  return t;
}
// Async copy of 16 bytes/lane from global to LDS (ASYNCcnt-tracked, CDNA5).
__device__ __forceinline__ void async_b128(unsigned lds_byte_off, const void* gaddr) {
  asm volatile("global_load_async_to_lds_b128 %0, %1, off"
               :: "v"(lds_byte_off), "v"((unsigned long long)(uintptr_t)gaddr)
               : "memory");
}

// ---------------- fp32 -> bf16 cast (4 elems/thread) ----------------
__global__ void cast_f32_bf16(const float* __restrict__ in, __bf16* __restrict__ out, int n) {
  int i = (blockIdx.x * 256 + threadIdx.x) * 4;
  if (i < n) {
    float4 f = *(const float4*)(in + i);
    v4bf o = { (__bf16)f.x, (__bf16)f.y, (__bf16)f.z, (__bf16)f.w };
    *(v4bf*)(out + i) = o;
  }
}

// ---------------- QKV projection: [4096,1024]x[1024,3072] ----------------
// One wave -> 32x64 tile (8 accumulators, A-frags reused across 4 B columns).
// Q,K stored [B,H,S,Hd]; V stored transposed [B,H,Hd,S].
__device__ __forceinline__ void store_qkv_frag(__bf16* __restrict__ Q, __bf16* __restrict__ K,
                                               __bf16* __restrict__ VT,
                                               int e, float bv, v8f c, int rowbase, int lhalf) {
  int part = e >> 10;                 // 0=q 1=k 2=v
  int eh = e & 1023;
  int h = eh >> 6, hd = eh & 63;
#pragma unroll
  for (int v = 0; v < 8; v++) {
    int row = rowbase + v + 8 * lhalf;
    int b_ = row >> 11, s = row & (SS - 1);
    size_t bh = (size_t)(b_ * NH + h);
    __bf16 val = (__bf16)(c[v] + bv);
    if      (part == 0) Q [(bh * SS + s) * HD + hd] = val;
    else if (part == 1) K [(bh * SS + s) * HD + hd] = val;
    else                VT[(bh * HD + hd) * SS + s] = val;
  }
}

__global__ void qkv_gemm(const __bf16* __restrict__ X, const __bf16* __restrict__ W,
                         const float* __restrict__ bias,
                         __bf16* __restrict__ Q, __bf16* __restrict__ K,
                         __bf16* __restrict__ VT) {
  const int NT = 3 * EMB / 64;                      // 48 tiles along N
  int wid   = blockIdx.x * 4 + (threadIdx.x >> 5);
  int lane  = threadIdx.x & 31;
  int lhalf = lane >> 4, lcol = lane & 15;
  int tn = wid % NT, tm = wid / NT;

  int akb = lhalf * 8;                              // A: K groups
  int bkb = lhalf * 8;                              // B: K groups

  const __bf16* ap0 = X + (size_t)(tm * 32 + lcol) * EMB;
  const __bf16* ap1 = ap0 + (size_t)16 * EMB;
  const __bf16* bp[4];
  int e[4];
#pragma unroll
  for (int g = 0; g < 4; g++) {
    e[g]  = tn * 64 + g * 16 + lcol;
    bp[g] = W + (size_t)e[g] * EMB;
  }

  v8f acc[8] = {};                                  // [g][rowblock]
#pragma unroll 2
  for (int k0 = 0; k0 < EMB; k0 += 32) {
    __builtin_prefetch(ap0 + k0 + 256, 0, 0);       // global_prefetch_b8
    v16bf a0 = join16(ld8(ap0 + k0 + akb), ld8(ap0 + k0 + akb + 16));
    v16bf a1 = join16(ld8(ap1 + k0 + akb), ld8(ap1 + k0 + akb + 16));
#pragma unroll
    for (int g = 0; g < 4; g++) {
      v16bf b = join16(ld8(bp[g] + k0 + bkb), ld8(bp[g] + k0 + bkb + 16));
      acc[2 * g]     = wmma_bf16(a0, b, acc[2 * g]);
      acc[2 * g + 1] = wmma_bf16(a1, b, acc[2 * g + 1]);
    }
  }

#pragma unroll
  for (int g = 0; g < 4; g++) {
    float bv = bias[e[g]];
    store_qkv_frag(Q, K, VT, e[g], bv, acc[2 * g],     tm * 32,      lhalf);
    store_qkv_frag(Q, K, VT, e[g], bv, acc[2 * g + 1], tm * 32 + 16, lhalf);
  }
}

// ---------------- Flash attention ----------------
// Block = 4 waves sharing one (b,h); each wave owns a 16-row q tile.
// K/V 32-key tiles staged in LDS once per block via async-to-LDS.
__global__ void attn_kernel(const __bf16* __restrict__ Q, const __bf16* __restrict__ Kb,
                            const __bf16* __restrict__ VT, __bf16* __restrict__ Y) {
  __shared__ __bf16 sK[32 * 64];                   // 32 keys x 64 d   (4KB)
  __shared__ __bf16 sV[64 * 32];                   // 64 hd  x 32 keys (4KB)
  __shared__ __bf16 sP[4][16 * 32];                // per-wave prob tile

  int tid   = threadIdx.x;
  int w     = tid >> 5;
  int lane  = tid & 31;
  int lhalf = lane >> 4, lcol = lane & 15;
  int bh    = blockIdx.x >> 5;                     // 32 blocks per (b,h)
  int qt    = (blockIdx.x & 31) * 4 + w;

  // Q fragments (A layout), held in registers for the whole pass
  const __bf16* qp = Q + ((size_t)bh * SS + qt * 16 + lcol) * HD;
  int akb = lhalf * 16;
  v16bf aq0 = join16(ld8(qp + akb),      ld8(qp + akb + 8));      // d 0..31
  v16bf aq1 = join16(ld8(qp + 32 + akb), ld8(qp + 32 + akb + 8)); // d 32..63

  v8f o0 = {}, o1 = {}, o2 = {}, o3 = {};
  float mrow[8], lrow[8];
#pragma unroll
  for (int v = 0; v < 8; v++) { mrow[v] = -1e30f; lrow[v] = 0.f; }

  const __bf16* kbase = Kb + (size_t)bh * SS * HD;
  const __bf16* vbase = VT + (size_t)bh * HD * SS;
  unsigned skB = (unsigned)(uintptr_t)&sK[0];      // flat low 32 bits = LDS byte addr
  unsigned svB = (unsigned)(uintptr_t)&sV[0];
  int vr = tid >> 2, vo = (tid & 3) * 8;           // V-tile copy: row, col-chunk

  for (int k0 = 0; k0 < SS; k0 += 32) {
    // ---- stage K tile (contiguous 4KB) and V tile into LDS, async ----
    const __bf16* kg = kbase + (size_t)k0 * HD;
    async_b128(skB + tid * 16,         kg + tid * 8);
    async_b128(skB + (tid + 128) * 16, kg + (tid + 128) * 8);
    async_b128(svB + (vr * 32 + vo) * 2,        vbase + (size_t)vr * SS + k0 + vo);
    async_b128(svB + ((vr + 32) * 32 + vo) * 2, vbase + (size_t)(vr + 32) * SS + k0 + vo);
    asm volatile("s_wait_asynccnt 0x0" ::: "memory");
    __syncthreads();

    // ---- scores for 32 keys (B frags from LDS), K-dim = 64 -> 2 wmma/tile ----
    const __bf16* skp0 = sK + (size_t)lcol * HD + lhalf * 16;        // keys 0..15
    const __bf16* skp1 = sK + (size_t)(16 + lcol) * HD + lhalf * 16; // keys 16..31
    v8f s0 = {}, s1 = {};
    s0 = wmma_bf16(aq0, join16(ld8(skp0),      ld8(skp0 + 8)),  s0);
    s0 = wmma_bf16(aq1, join16(ld8(skp0 + 32), ld8(skp0 + 40)), s0);
    s1 = wmma_bf16(aq0, join16(ld8(skp1),      ld8(skp1 + 8)),  s1);
    s1 = wmma_bf16(aq1, join16(ld8(skp1 + 32), ld8(skp1 + 40)), s1);

    // ---- online softmax (rows live in 16-lane halves at fixed acc index) ----
#pragma unroll
    for (int v = 0; v < 8; v++) {
      float a = s0[v] * 0.125f, b = s1[v] * 0.125f;    // 1/sqrt(64)
      float rm = rmax16(fmaxf(a, b));
      float mn = fmaxf(mrow[v], rm);
      float al = __builtin_amdgcn_exp2f((mrow[v] - mn) * LOG2E);
      mrow[v] = mn;
      float p0 = __builtin_amdgcn_exp2f((a - mn) * LOG2E);
      float p1 = __builtin_amdgcn_exp2f((b - mn) * LOG2E);
      lrow[v] = lrow[v] * al + rsum16(p0 + p1);
      o0[v] *= al; o1[v] *= al; o2[v] *= al; o3[v] *= al;
      int r = v + 8 * lhalf;
      sP[w][r * 32 + lcol]      = (__bf16)p0;          // C-layout -> row-major LDS
      sP[w][r * 32 + 16 + lcol] = (__bf16)p1;
    }
    asm volatile("s_wait_dscnt 0x0" ::: "memory");

    // ---- re-read prob tile in A-fragment layout (16x32, K=keys) ----
    v16bf apf = join16(*(const v8bf*)&sP[w][lcol * 32 + akb],
                       *(const v8bf*)&sP[w][lcol * 32 + akb + 8]);

    // ---- attn @ V : N = Hd = 64 -> 4 wmma, B frags from LDS ----
    const __bf16* vp0 = sV + (size_t)(0  + lcol) * 32 + lhalf * 16;
    const __bf16* vp1 = sV + (size_t)(16 + lcol) * 32 + lhalf * 16;
    const __bf16* vp2 = sV + (size_t)(32 + lcol) * 32 + lhalf * 16;
    const __bf16* vp3 = sV + (size_t)(48 + lcol) * 32 + lhalf * 16;
    o0 = wmma_bf16(apf, join16(ld8(vp0), ld8(vp0 + 8)), o0);
    o1 = wmma_bf16(apf, join16(ld8(vp1), ld8(vp1 + 8)), o1);
    o2 = wmma_bf16(apf, join16(ld8(vp2), ld8(vp2 + 8)), o2);
    o3 = wmma_bf16(apf, join16(ld8(vp3), ld8(vp3 + 8)), o3);

    __syncthreads();                                 // protect sK/sV before overwrite
  }

  // ---- normalize and write Y in [B,S,D] bf16 ----
  int b_ = bh >> 4, h = bh & 15;
#pragma unroll
  for (int v = 0; v < 8; v++) {
    float inv = 1.0f / lrow[v];
    int s = qt * 16 + v + 8 * lhalf;
    size_t base = ((size_t)b_ * SS + s) * EMB + h * HD + lcol;
    Y[base +  0] = (__bf16)(o0[v] * inv);
    Y[base + 16] = (__bf16)(o1[v] * inv);
    Y[base + 32] = (__bf16)(o2[v] * inv);
    Y[base + 48] = (__bf16)(o3[v] * inv);
  }
}

// ---------------- Output projection: [4096,1024]x[1024,1024] -> fp32 ----------------
__global__ void out_gemm(const __bf16* __restrict__ Yb, const __bf16* __restrict__ W,
                         const float* __restrict__ bias, float* __restrict__ Out) {
  const int NT = EMB / 64;                          // 16 tiles along N
  int wid   = blockIdx.x * 4 + (threadIdx.x >> 5);
  int lane  = threadIdx.x & 31;
  int lhalf = lane >> 4, lcol = lane & 15;
  int tn = wid % NT, tm = wid / NT;

  int akb = lhalf * 8;
  int bkb = lhalf * 8;

  const __bf16* ap0 = Yb + (size_t)(tm * 32 + lcol) * EMB;
  const __bf16* ap1 = ap0 + (size_t)16 * EMB;
  const __bf16* bp[4];
  int e[4];
#pragma unroll
  for (int g = 0; g < 4; g++) {
    e[g]  = tn * 64 + g * 16 + lcol;
    bp[g] = W + (size_t)e[g] * EMB;
  }

  v8f acc[8] = {};
#pragma unroll 2
  for (int k0 = 0; k0 < EMB; k0 += 32) {
    __builtin_prefetch(ap0 + k0 + 256, 0, 0);
    v16bf a0 = join16(ld8(ap0 + k0 + akb), ld8(ap0 + k0 + akb + 16));
    v16bf a1 = join16(ld8(ap1 + k0 + akb), ld8(ap1 + k0 + akb + 16));
#pragma unroll
    for (int g = 0; g < 4; g++) {
      v16bf b = join16(ld8(bp[g] + k0 + bkb), ld8(bp[g] + k0 + bkb + 16));
      acc[2 * g]     = wmma_bf16(a0, b, acc[2 * g]);
      acc[2 * g + 1] = wmma_bf16(a1, b, acc[2 * g + 1]);
    }
  }

#pragma unroll
  for (int g = 0; g < 4; g++) {
    float bv = bias[e[g]];
#pragma unroll
    for (int v = 0; v < 8; v++) {
      int r0 = tm * 32 + v + 8 * lhalf;
      Out[(size_t)r0 * EMB + e[g]]        = acc[2 * g][v] + bv;
      Out[(size_t)(r0 + 16) * EMB + e[g]] = acc[2 * g + 1][v] + bv;
    }
  }
}

extern "C" void kernel_launch(void* const* d_in, const int* in_sizes, int n_in,
                              void* d_out, int out_size, void* d_ws, size_t ws_size,
                              hipStream_t stream) {
  const float* x     = (const float*)d_in[0];   // [2,2048,1024]
  const float* qkv_w = (const float*)d_in[1];   // [3072,1024]
  const float* qkv_b = (const float*)d_in[2];   // [3072]
  const float* out_w = (const float*)d_in[3];   // [1024,1024]
  const float* out_b = (const float*)d_in[4];   // [1024]
  float* out = (float*)d_out;

  char* ws = (char*)d_ws;
  size_t off = 0;
  __bf16* xb   = (__bf16*)(ws + off); off += (size_t)MROWS * EMB * 2;       // 8 MB
  __bf16* wqkv = (__bf16*)(ws + off); off += (size_t)3 * EMB * EMB * 2;     // 6 MB
  __bf16* wo   = (__bf16*)(ws + off); off += (size_t)EMB * EMB * 2;         // 2 MB
  __bf16* qb   = (__bf16*)(ws + off); off += (size_t)MROWS * EMB * 2;       // 8 MB
  __bf16* kb   = (__bf16*)(ws + off); off += (size_t)MROWS * EMB * 2;       // 8 MB
  __bf16* vtb  = (__bf16*)(ws + off); off += (size_t)MROWS * EMB * 2;       // 8 MB
  __bf16* yb   = (__bf16*)(ws + off); off += (size_t)MROWS * EMB * 2;       // 8 MB
  (void)ws_size; (void)n_in; (void)in_sizes; (void)out_size;

  // 1) casts to bf16
  {
    int n = MROWS * EMB;
    cast_f32_bf16<<<n / 1024, 256, 0, stream>>>(x, xb, n);
    n = 3 * EMB * EMB;
    cast_f32_bf16<<<n / 1024, 256, 0, stream>>>(qkv_w, wqkv, n);
    n = EMB * EMB;
    cast_f32_bf16<<<n / 1024, 256, 0, stream>>>(out_w, wo, n);
  }
  // 2) fused QKV projection: (4096/32)x(3072/64) = 6144 wave-tiles, 4 waves/block
  qkv_gemm<<<(MROWS / 32) * (3 * EMB / 64) / 4, 128, 0, stream>>>(
      xb, wqkv, qkv_b, qb, kb, vtb);
  // 3) flash attention: 32 (b,h) x 128 q-tiles, 4 waves/block
  attn_kernel<<<(BB * NH * (SS / 16)) / 4, 128, 0, stream>>>(qb, kb, vtb, yb);
  // 4) output projection: (4096/32)x(1024/64) = 2048 wave-tiles
  out_gemm<<<(MROWS / 32) * (EMB / 64) / 4, 128, 0, stream>>>(yb, wo, out_b, out);
}
